// MultiHeadSelfAttention_3453153706729
// MI455X (gfx1250) — compile-verified, matrix operand order checked by reading, not run on hardware
//
#include <hip/hip_runtime.h>

typedef __attribute__((ext_vector_type(16))) _Float16 v16h;
typedef __attribute__((ext_vector_type(8)))  _Float16 v8h;
typedef __attribute__((ext_vector_type(8)))  float    v8f;

#define DIM      1024
#define HEADS    16
#define HD       64
#define BATCH    512
#define SEQ      65
#define ROWS     (BATCH * SEQ)      // 33280 = 16*2080
#define QKV_COLS (3 * DIM)          // 3072

// ---------------------------------------------------------------------------
// One-shot f32 -> f16 conversion (4 elements / thread, vectorized).
// ---------------------------------------------------------------------------
__global__ __launch_bounds__(256) void cvt_f32_f16(const float* __restrict__ src,
                                                   _Float16* __restrict__ dst,
                                                   long long n4) {
    long long i = (long long)blockIdx.x * 256 + threadIdx.x;
    if (i >= n4) return;
    float4 f = ((const float4*)src)[i];
    _Float16 h[4] = {(_Float16)f.x, (_Float16)f.y, (_Float16)f.z, (_Float16)f.w};
    *(((ulong1*)dst) + i) = *(ulong1*)h;   // 8-byte store of 4 halfs
}

// ---------------------------------------------------------------------------
// GEMM: C[M,Ncols] = A[M,K](f16) * W[Ncols,K](f16)^T, per-wave 16x128 tile,
// f32 accumulation. OUT_F16 selects f16 (intermediate) or f32 (final) store.
// ---------------------------------------------------------------------------
template <bool OUT_F16>
__global__ __launch_bounds__(256) void gemm_h(const _Float16* __restrict__ A,
                                              const _Float16* __restrict__ W,
                                              void* __restrict__ Cv,
                                              int K, int Ncols) {
    const int lane = threadIdx.x & 31;
    const int wave = threadIdx.x >> 5;
    const int l16  = lane & 15;
    const int hs   = lane >> 4;
    const int rowBase = blockIdx.x * 16;
    const int colBase = blockIdx.y * 1024 + wave * 128;

    const _Float16* arow  = A + (size_t)(rowBase + l16) * K + hs * 16;
    const _Float16* wbase = W + (size_t)(colBase + l16) * K + hs * 16;

    v8f acc[8] = {};
    for (int k = 0; k < K; k += 32) {
        v16h a = *(const v16h*)(arow + k);
#pragma unroll
        for (int c = 0; c < 8; ++c) {
            v16h b = *(const v16h*)(wbase + (size_t)(c * 16) * K + k);
            acc[c] = __builtin_amdgcn_wmma_f32_16x16x32_f16(false, a, false, b,
                                                            (short)0, acc[c], false, false);
        }
    }
#pragma unroll
    for (int v = 0; v < 8; ++v) {
        int row = rowBase + v + 8 * hs;
        if (OUT_F16) {
            _Float16* crow = (_Float16*)Cv + (size_t)row * Ncols + colBase + l16;
#pragma unroll
            for (int c = 0; c < 8; ++c) crow[c * 16] = (_Float16)acc[c][v];
        } else {
            float* crow = (float*)Cv + (size_t)row * Ncols + colBase + l16;
#pragma unroll
            for (int c = 0; c < 8; ++c) crow[c * 16] = acc[c][v];
        }
    }
}

// ---------------------------------------------------------------------------
// Single-pass vectorized LayerNorm over one 64-element head row (registers
// hold the row as 8x v8h); writes f16 (pre-scaled) to LDS in 16B chunks.
// ---------------------------------------------------------------------------
__device__ __forceinline__ void ln_row(const _Float16* __restrict__ src,
                                       const float* __restrict__ w,
                                       const float* __restrict__ bp,
                                       _Float16* __restrict__ dst,
                                       float postscale) {
    v8h c[8];
#pragma unroll
    for (int j = 0; j < 8; ++j) c[j] = ((const v8h*)src)[j];
    float s = 0.f, ss = 0.f;
#pragma unroll
    for (int j = 0; j < 8; ++j)
#pragma unroll
        for (int i = 0; i < 8; ++i) { float f = (float)c[j][i]; s += f; ss += f * f; }
    float mu  = s * (1.0f / HD);
    float var = ss * (1.0f / HD) - mu * mu;
    float inv = rsqrtf(var + 1e-5f);
#pragma unroll
    for (int j = 0; j < 8; ++j) {
        v8h o;
#pragma unroll
        for (int i = 0; i < 8; ++i) {
            float f = (float)c[j][i];
            o[i] = (_Float16)(((f - mu) * inv * w[j * 8 + i] + bp[j * 8 + i]) * postscale);
        }
        ((v8h*)dst)[j] = o;
    }
}

// ---------------------------------------------------------------------------
// One workgroup per (b,h): LN(q,k) -> S = QK^T + bias -> softmax -> O = P V.
// ---------------------------------------------------------------------------
__global__ __launch_bounds__(256) void attn_kernel(const _Float16* __restrict__ qkv,
                                                   const float* __restrict__ qn_w,
                                                   const float* __restrict__ qn_b,
                                                   const float* __restrict__ kn_w,
                                                   const float* __restrict__ kn_b,
                                                   const float* __restrict__ bias,
                                                   const float* __restrict__ bias_scale,
                                                   _Float16* __restrict__ aout) {
    __shared__ __align__(128) _Float16 Qs[80][64];       // normalized Q, pad rows = 0
    __shared__ __align__(128) _Float16 Ks[80][64];       // normalized K
    __shared__ __align__(128) _Float16 Vt[64][96];       // V transposed, pad cols = 0
    __shared__ __align__(128) _Float16 Ps[8][16][96];    // per-wave P tiles, pad K = 0
    __shared__ __align__(128) float    Bs[80][80];       // pre-scaled bias tile, pads = 0

    const int t = threadIdx.x;
    const int b = blockIdx.x >> 4;
    const int h = blockIdx.x & 15;
    const _Float16* qb = qkv + (size_t)b * SEQ * QKV_COLS + h * HD;
    const _Float16* kb = qb + DIM;
    const _Float16* vb = qb + 2 * DIM;

    for (int e = t; e < 80 * 64; e += 256) { (&Qs[0][0])[e] = (_Float16)0.f; (&Ks[0][0])[e] = (_Float16)0.f; }
    for (int e = t; e < 64 * 96; e += 256) (&Vt[0][0])[e] = (_Float16)0.f;
    for (int e = t; e < 8 * 16 * 96; e += 256) (&Ps[0][0][0])[e] = (_Float16)0.f;
    for (int e = t; e < 80 * 80; e += 256) (&Bs[0][0])[e] = 0.f;
    __syncthreads();

    // Stage V (transposed) and the pre-scaled bias tile; LayerNorm q/k rows.
    const float bsc = bias_scale[0];
    for (int e = t; e < SEQ * SEQ; e += 256) {
        int r = e / SEQ, c = e - r * SEQ;
        Bs[r][c] = bias[((size_t)h * SEQ + r) * SEQ + c] * bsc;
    }
    for (int e = t; e < SEQ * HD; e += 256) {
        int m = e >> 6, d = e & 63;
        Vt[d][m] = vb[(size_t)m * QKV_COLS + d];
    }
    if (t < SEQ) {
        ln_row(qb + (size_t)t * QKV_COLS, qn_w, qn_b, &Qs[t][0], 0.125f);   // * hd^-0.5
    } else if (t >= 128 && t < 128 + SEQ) {
        int r = t - 128;
        ln_row(kb + (size_t)r * QKV_COLS, kn_w, kn_b, &Ks[r][0], 1.0f);
    }
    __syncthreads();

    const int wave = t >> 5, lane = t & 31, l16 = lane & 15, hs = lane >> 4;
    if (wave < 5) {                       // 5 row tiles cover 65 (padded to 80) rows
        v8f S[5] = {};
#pragma unroll
        for (int ks = 0; ks < HD; ks += 32) {
            v16h a = *(const v16h*)&Qs[wave * 16 + l16][ks + hs * 16];
#pragma unroll
            for (int nt = 0; nt < 5; ++nt) {
                v16h bf = *(const v16h*)&Ks[nt * 16 + l16][ks + hs * 16];
                S[nt] = __builtin_amdgcn_wmma_f32_16x16x32_f16(false, a, false, bf,
                                                               (short)0, S[nt], false, false);
            }
        }
#pragma unroll
        for (int v = 0; v < 8; ++v) {
            const int row = wave * 16 + v + 8 * hs;     // < 80, always in-bounds for Bs
            float ex[5];
            float m = -3.0e38f;
#pragma unroll
            for (int nt = 0; nt < 5; ++nt) {
                const int col = nt * 16 + l16;
                float sv = S[nt][v] + Bs[row][col];     // ds_load_b32, pads contribute 0
                ex[nt] = (col < SEQ) ? sv : -3.0e38f;
                m = fmaxf(m, ex[nt]);
            }
#pragma unroll
            for (int off = 1; off < 16; off <<= 1) m = fmaxf(m, __shfl_xor(m, off, 32));
            float sum = 0.f;
#pragma unroll
            for (int nt = 0; nt < 5; ++nt) {
                float p = __expf(ex[nt] - m);           // masked cols underflow to 0
                ex[nt] = p;
                sum += p;
            }
#pragma unroll
            for (int off = 1; off < 16; off <<= 1) sum += __shfl_xor(sum, off, 32);
            float inv = 1.0f / sum;
#pragma unroll
            for (int nt = 0; nt < 5; ++nt)
                Ps[wave][v + 8 * hs][nt * 16 + l16] = (_Float16)(ex[nt] * inv);
        }
        // O = P * V  (K = 80 padded to 96; pads are zero)
        v8f O[4] = {};
#pragma unroll
        for (int ks = 0; ks < 96; ks += 32) {
            v16h a = *(const v16h*)&Ps[wave][l16][ks + hs * 16];
#pragma unroll
            for (int ct = 0; ct < 4; ++ct) {
                v16h bf = *(const v16h*)&Vt[ct * 16 + l16][ks + hs * 16];
                O[ct] = __builtin_amdgcn_wmma_f32_16x16x32_f16(false, a, false, bf,
                                                               (short)0, O[ct], false, false);
            }
        }
#pragma unroll
        for (int ct = 0; ct < 4; ++ct) {
#pragma unroll
            for (int v = 0; v < 8; ++v) {
                int n = wave * 16 + v + 8 * hs;
                if (n < SEQ)
                    aout[((size_t)b * SEQ + n) * DIM + h * HD + ct * 16 + l16] = (_Float16)O[ct][v];
            }
        }
    }
}

extern "C" void kernel_launch(void* const* d_in, const int* in_sizes, int n_in,
                              void* d_out, int out_size, void* d_ws, size_t ws_size,
                              hipStream_t stream) {
    (void)in_sizes; (void)n_in; (void)out_size; (void)ws_size;
    const float* x      = (const float*)d_in[0];
    const float* w_qkv  = (const float*)d_in[1];
    const float* w_proj = (const float*)d_in[2];
    const float* qn_w   = (const float*)d_in[3];
    const float* qn_b   = (const float*)d_in[4];
    const float* kn_w   = (const float*)d_in[5];
    const float* kn_b   = (const float*)d_in[6];
    const float* bias   = (const float*)d_in[7];
    const float* bscale = (const float*)d_in[8];

    // Workspace layout (all f16):
    _Float16* xh     = (_Float16*)d_ws;                         // [33280,1024]  68 MB
    _Float16* wqkvh  = xh     + (size_t)ROWS * DIM;             // [3072,1024]    6 MB
    _Float16* wprojh = wqkvh  + (size_t)QKV_COLS * DIM;         // [1024,1024]    2 MB
    _Float16* qkv    = wprojh + (size_t)DIM * DIM;              // [33280,3072] 204 MB
    _Float16* aout   = qkv    + (size_t)ROWS * QKV_COLS;        // [33280,1024]  68 MB
    float* out = (float*)d_out;                                 // [33280,1024] f32

    // 1) one-shot f32->f16 of activations + weights (x_h then stays L2-resident)
    long long nx = (long long)ROWS * DIM / 4;
    long long nq = (long long)QKV_COLS * DIM / 4;
    long long np = (long long)DIM * DIM / 4;
    cvt_f32_f16<<<dim3((unsigned)((nx + 255) / 256)), 256, 0, stream>>>(x, xh, nx);
    cvt_f32_f16<<<dim3((unsigned)((nq + 255) / 256)), 256, 0, stream>>>(w_qkv, wqkvh, nq);
    cvt_f32_f16<<<dim3((unsigned)((np + 255) / 256)), 256, 0, stream>>>(w_proj, wprojh, np);

    // 2) QKV projection: [33280,1024] x [1024,3072] -> f16
    gemm_h<true><<<dim3(ROWS / 16, QKV_COLS / 1024), 256, 0, stream>>>(xh, wqkvh, qkv, DIM, QKV_COLS);

    // 3) fused LN + attention + softmax + PV per (b,h)
    attn_kernel<<<dim3(BATCH * HEADS), 256, 0, stream>>>(qkv, qn_w, qn_b, kn_w, kn_b, bias, bscale, aout);

    // 4) output projection: [33280,1024] x [1024,1024] -> f32
    gemm_h<false><<<dim3(ROWS / 16, DIM / 1024), 256, 0, stream>>>(aout, wprojh, out, DIM, DIM);
}
